// CorefContrastiveLoss_88046829568104
// MI455X (gfx1250) — matrix operand order repaired and checked by reading, not compiled, not run on hardware
//
#include <hip/hip_runtime.h>
#include <hip/hip_bf16.h>
#include <math.h>

// ---------------------------------------------------------------------------
// CorefContrastiveLoss on MI455X (gfx1250, wave32)
//   x = normalize_rows(embs) -> bf16 (fits in 192MB L2: 16MB)
//   sim = (x x^T) * 5.0  via v_wmma_f32_16x16x32_bf16
//   fused log-softmax epilogue with FIXED max = 5.0 (|cos|<=1 -> s in [-5,5])
//   loss = -(1/n) sum_i [ A_i - B_i * (5 + log S_i) ],
//     S_i = sum_{j!=i} exp(s_ij - 5), A_i = sum lbl*s, B_i = sum lbl
// ---------------------------------------------------------------------------

typedef __attribute__((ext_vector_type(16))) __bf16 v16bf;
typedef __attribute__((ext_vector_type(8)))  __bf16 v8bf;
typedef __attribute__((ext_vector_type(4)))  __bf16 v4bf;
typedef __attribute__((ext_vector_type(8)))  float  v8f;

#define N_TOT     8192
#define DIM       1024
#define TEMP_INV  5.0f
#define LDS_STRIDE 1032              // 1024 + 8 bf16 pad -> conflict-free b128 LDS reads
#define SMEM_A_BYTES (32 * LDS_STRIDE * 2)   // 66048
#define SMEM_RED_FLOATS (8 * 32)             // per-wave, per-row partials
#define SMEM_TOTAL (SMEM_A_BYTES + 3 * SMEM_RED_FLOATS * 4)  // 69120 B

__global__ void coref_init_out(float* out) {
    if (threadIdx.x == 0 && blockIdx.x == 0) out[0] = 0.0f;
}

// One block (256 threads) per row: L2-norm, clamp, write bf16.
__global__ void __launch_bounds__(256) coref_normalize_bf16(
        const float* __restrict__ embs, __bf16* __restrict__ xbf) {
    const int row = blockIdx.x;
    const int t   = threadIdx.x;
    const float4 va = ((const float4*)(embs + (size_t)row * DIM))[t];
    float ss = va.x * va.x + va.y * va.y + va.z * va.z + va.w * va.w;
    #pragma unroll
    for (int m = 1; m < 32; m <<= 1) ss += __shfl_xor(ss, m, 32);
    __shared__ float red[8];
    const int wave = t >> 5, lane = t & 31;
    if (lane == 0) red[wave] = ss;
    __syncthreads();
    float tot = 0.0f;
    #pragma unroll
    for (int w = 0; w < 8; ++w) tot += red[w];
    const float inv = 1.0f / fmaxf(sqrtf(tot), 1e-8f);
    v4bf o;
    o.x = (__bf16)(va.x * inv);
    o.y = (__bf16)(va.y * inv);
    o.z = (__bf16)(va.z * inv);
    o.w = (__bf16)(va.w * inv);
    ((v4bf*)(xbf + (size_t)row * DIM))[t] = o;
}

// One workgroup per 32-row M block. 8 waves split the 256 N-tile-pairs.
// Per wave per N-pair: K-loop of 32 steps, 2 A-frags (LDS) x 2 B-frags (L2)
// -> 4 WMMAs/step; fused softmax epilogue reads labels once.
__global__ void __launch_bounds__(256, 1) coref_loss_kernel(
        const __bf16* __restrict__ xbf,
        const float*  __restrict__ labels,
        float*        __restrict__ out) {
    extern __shared__ char smem[];
    __bf16* ldsA = (__bf16*)smem;
    float*  redS = (float*)(smem + SMEM_A_BYTES);
    float*  redA = redS + SMEM_RED_FLOATS;
    float*  redB = redA + SMEM_RED_FLOATS;

    const int mBase = blockIdx.x * 32;
    const int t     = threadIdx.x;
    const int wave  = t >> 5;
    const int lane  = t & 31;
    const int hh    = lane >> 4;   // lane half (K-half selector per WMMA layouts)
    const int l16   = lane & 15;   // M-row (A frag) / N-col (B frag)

    // Stage 32 rows of x (64KB) into padded LDS.
    #pragma unroll
    for (int c = 0; c < 16; ++c) {
        const int idx  = t + c * 256;       // 4096 chunks of 8 bf16
        const int row  = idx >> 7;          // 128 chunks per row
        const int col8 = idx & 127;
        v8bf d = *(const v8bf*)(xbf + (size_t)(mBase + row) * DIM + col8 * 8);
        *(v8bf*)(&ldsA[row * LDS_STRIDE + col8 * 8]) = d;
    }
    __syncthreads();

    // Per-(lane,row) partial sums; fixed max = 5.0, so no online rescaling.
    float accS[2][8], accA[2][8], accB[2][8];
    #pragma unroll
    for (int mt = 0; mt < 2; ++mt)
        #pragma unroll
        for (int v = 0; v < 8; ++v) { accS[mt][v] = 0.f; accA[mt][v] = 0.f; accB[mt][v] = 0.f; }

    const __bf16* aRow0 = &ldsA[(l16)      * LDS_STRIDE];
    const __bf16* aRow1 = &ldsA[(16 + l16) * LDS_STRIDE];

    for (int tp = wave; tp < N_TOT / 32; tp += 8) {
        const int jBase = tp * 32;
        const __bf16* bRow0 = xbf + (size_t)(jBase + l16)      * DIM + hh * 16;
        const __bf16* bRow1 = xbf + (size_t)(jBase + 16 + l16) * DIM + hh * 16;

        v8f c00 = {}, c01 = {}, c10 = {}, c11 = {};
        #pragma unroll 4
        for (int k0 = 0; k0 < DIM; k0 += 32) {
            union { v16bf v; v8bf h[2]; } a0, a1, b0, b1;
            // A 16x32 bf16: lane half 0 holds K 0..15, half 1 K 16..31
            a0.h[0] = *(const v8bf*)(aRow0 + k0 + hh * 16);
            a0.h[1] = *(const v8bf*)(aRow0 + k0 + hh * 16 + 8);
            a1.h[0] = *(const v8bf*)(aRow1 + k0 + hh * 16);
            a1.h[1] = *(const v8bf*)(aRow1 + k0 + hh * 16 + 8);
            // B 32x16 bf16: lane half 0 holds K 0..15, half 1 K 16..31 of column l16
            b0.v = *(const v16bf*)(bRow0 + k0);
            b1.v = *(const v16bf*)(bRow1 + k0);
            c00 = __builtin_amdgcn_wmma_f32_16x16x32_bf16(false, a0.v, false, b0.v, (short)0, c00, false, false);
            c01 = __builtin_amdgcn_wmma_f32_16x16x32_bf16(false, a0.v, false, b1.v, (short)0, c01, false, false);
            c10 = __builtin_amdgcn_wmma_f32_16x16x32_bf16(false, a1.v, false, b0.v, (short)0, c10, false, false);
            c11 = __builtin_amdgcn_wmma_f32_16x16x32_bf16(false, a1.v, false, b1.v, (short)0, c11, false, false);
        }

        // Fused epilogue: C layout -> row = v + 8*hh (per tile), col = l16.
        #pragma unroll
        for (int mt = 0; mt < 2; ++mt) {
            #pragma unroll
            for (int nt = 0; nt < 2; ++nt) {
                const v8f cc = (mt == 0) ? (nt == 0 ? c00 : c01) : (nt == 0 ? c10 : c11);
                const int j = jBase + nt * 16 + l16;
                #pragma unroll
                for (int v = 0; v < 8; ++v) {
                    const int i = mBase + mt * 16 + hh * 8 + v;
                    if (i != j) {
                        const float s   = cc[v] * TEMP_INV;
                        const float lbl = labels[(size_t)i * N_TOT + j];
                        accS[mt][v] += __expf(s - 5.0f);
                        accA[mt][v] += lbl * s;
                        accB[mt][v] += lbl;
                    }
                }
            }
        }
    }

    // Reduce across the 16 lanes sharing each row, stash per-wave partials.
    #pragma unroll
    for (int mt = 0; mt < 2; ++mt) {
        #pragma unroll
        for (int v = 0; v < 8; ++v) {
            float S = accS[mt][v], A = accA[mt][v], B = accB[mt][v];
            #pragma unroll
            for (int msk = 1; msk < 16; msk <<= 1) {
                S += __shfl_xor(S, msk, 32);
                A += __shfl_xor(A, msk, 32);
                B += __shfl_xor(B, msk, 32);
            }
            if (l16 == 0) {
                const int r = mt * 16 + hh * 8 + v;   // row within WG block
                redS[wave * 32 + r] = S;
                redA[wave * 32 + r] = A;
                redB[wave * 32 + r] = B;
            }
        }
    }
    __syncthreads();

    // Wave 0: combine 8 waves per row, per-row loss, reduce, atomicAdd scalar.
    if (t < 32) {
        float S = 0.f, A = 0.f, B = 0.f;
        #pragma unroll
        for (int w = 0; w < 8; ++w) {
            S += redS[w * 32 + t];
            A += redA[w * 32 + t];
            B += redB[w * 32 + t];
        }
        float contrib = A - B * (5.0f + __logf(S));   // lse = fixed_max + log S
        #pragma unroll
        for (int msk = 1; msk < 32; msk <<= 1) contrib += __shfl_xor(contrib, msk, 32);
        if (t == 0) atomicAdd(out, -contrib / (float)N_TOT);
    }
}

extern "C" void kernel_launch(void* const* d_in, const int* in_sizes, int n_in,
                              void* d_out, int out_size, void* d_ws, size_t ws_size,
                              hipStream_t stream) {
    const float* embs   = (const float*)d_in[0];   // (8192, 1024) fp32
    const float* labels = (const float*)d_in[1];   // (8192, 8192) fp32
    float* out = (float*)d_out;                    // scalar fp32
    __bf16* xbf = (__bf16*)d_ws;                   // 16 MB normalized bf16 x

    coref_init_out<<<1, 64, 0, stream>>>(out);
    coref_normalize_bf16<<<N_TOT, 256, 0, stream>>>(embs, xbf);
    coref_loss_kernel<<<N_TOT / 32, 256, SMEM_TOTAL, stream>>>(xbf, labels, out);
}
